// QuasiLSTM_70489003261961
// MI455X (gfx1250) — matrix operand order repaired and checked by reading, not compiled, not run on hardware
//
#include <hip/hip_runtime.h>

#define DIM   1024
#define BATCH 4
#define SEQ   8192
#define BL    (BATCH*SEQ)   // 32768 rows
#define NGATE 4
#define NC    128           // chunks per sequence
#define CL    64            // chunk length (NC*CL == SEQ)
#define KSTEPS (DIM/32)     // 32 K-iterations

typedef __attribute__((ext_vector_type(16))) __bf16 v16bf;
typedef __attribute__((ext_vector_type(8)))  __bf16 v8bf;
typedef __attribute__((ext_vector_type(4)))  __bf16 v4bf;
typedef __attribute__((ext_vector_type(8)))  float  v8f;

#if __has_builtin(__builtin_amdgcn_sched_barrier)
#define SCHED_FENCE() __builtin_amdgcn_sched_barrier(0)
#else
#define SCHED_FENCE() ((void)0)
#endif

#if __has_builtin(__builtin_amdgcn_sched_group_barrier)
// masks: 0x8 = MFMA/WMMA, 0x100 = DS read, 0x200 = DS write
#define SG(mask, size) __builtin_amdgcn_sched_group_barrier((mask), (size), 0)
#else
#define SG(mask, size) ((void)0)
#endif

// pipeline shape for one staged tile: 8 DS reads up front (A + 3 B pairs),
// then 1 WMMA : 2 DS reads x5, then 3 WMMAs, then the staging DS writes.
#define SCHED_TILE_COMPUTE()                                             \
    do {                                                                 \
        SG(0x100, 8);                                                    \
        SG(0x8, 1); SG(0x100, 2);                                        \
        SG(0x8, 1); SG(0x100, 2);                                        \
        SG(0x8, 1); SG(0x100, 2);                                        \
        SG(0x8, 1); SG(0x100, 2);                                        \
        SG(0x8, 1); SG(0x100, 2);                                        \
        SG(0x8, 3);                                                      \
    } while (0)
#define SCHED_TILE_FULL()                                                \
    do {                                                                 \
        SCHED_TILE_COMPUTE();                                            \
        SG(0x200, 4);                                                    \
    } while (0)

__device__ __forceinline__ float sigmoidf_(float v) {
    return 1.0f / (1.0f + __expf(-v));
}

// A fragment: lane = M row; K contiguous kb..kb+15, kb = 16*(lane>>4)
__device__ __forceinline__ v16bf load_afrag(const __bf16* p, int kb) {
    v8bf lo = *(const v8bf*)(p + kb);
    v8bf hi = *(const v8bf*)(p + kb + 8);
    return __builtin_shufflevector(lo, hi,
        0, 1, 2, 3, 4, 5, 6, 7, 8, 9, 10, 11, 12, 13, 14, 15);
}

// B fragment: lane = N col; K contiguous 0..15 (lanes 0-15) / 16..31 (lanes 16-31)
__device__ __forceinline__ v16bf load_bfrag(const __bf16* q, int kb2) {
    v8bf lo = *(const v8bf*)(q + kb2);
    v8bf hi = *(const v8bf*)(q + kb2 + 8);
    return __builtin_shufflevector(lo, hi,
        0, 1, 2, 3, 4, 5, 6, 7, 8, 9, 10, 11, 12, 13, 14, 15);
}

// ---------------- x (f32) -> bf16, contiguous -------------------------------
__global__ __launch_bounds__(256) void k_cvt_x(const float* __restrict__ x,
                                               __bf16* __restrict__ xb) {
    int i = (blockIdx.x * 256 + threadIdx.x) * 4;
    float4 v = *(const float4*)(x + i);
    v4bf o;
    o.x = (__bf16)v.x; o.y = (__bf16)v.y; o.z = (__bf16)v.z; o.w = (__bf16)v.w;
    *(v4bf*)(xb + i) = o;
}

// ---------------- W[k][n] f32 -> Wt[g*1024+n][k] bf16 (tiled transpose) -----
__global__ __launch_bounds__(256) void k_tr_w(const float* __restrict__ Wf,
                                              const float* __restrict__ Wi,
                                              const float* __restrict__ Wig,
                                              const float* __restrict__ Wog,
                                              __bf16* __restrict__ wt) {
    __shared__ float tile[32][33];
    const int g  = blockIdx.z;
    const float* W = (g == 0) ? Wf : (g == 1) ? Wi : (g == 2) ? Wig : Wog;
    const int nT = blockIdx.x * 32, kT = blockIdx.y * 32;
    const int tx = threadIdx.x & 31, ty = threadIdx.x >> 5;
#pragma unroll
    for (int i = 0; i < 4; ++i) {
        int k = ty + i * 8;
        tile[k][tx] = W[(size_t)(kT + k) * DIM + nT + tx];   // coalesced over n
    }
    __syncthreads();
#pragma unroll
    for (int i = 0; i < 4; ++i) {
        int n = ty + i * 8;
        wt[(size_t)(g * DIM + nT + n) * DIM + kT + tx] = (__bf16)tile[tx][n]; // coalesced over k
    }
}

// ---------------- fused 4-gate GEMM + activations ---------------------------
// block = 256 thr (8 waves). Block tile: 128 rows x 32 cols, for all 4 gates.
// Wave w owns rows w*16..w*16+15; 8 accumulators (4 gates x 2 n-subtiles).
// Double-buffered LDS, branch-free double-step loop, sched-group pipelining.

typedef __bf16 ldsrow_t[40];

// one compute pass over a staged 128x32 tile (8 wmmas, lookahead-2 B loads)
__device__ __forceinline__ void compute_tile(const ldsrow_t* lA, const ldsrow_t* lB,
                                             int ar, int br, int kb, int kb2,
                                             v8f acc[8]) {
    v16bf a  = load_afrag(&lA[ar][0], kb2);
    v16bf b0 = load_bfrag(&lB[br][0], kb2);
    v16bf b1 = load_bfrag(&lB[16 + br][0], kb2);
#pragma unroll
    for (int j = 0; j < 8; ++j) {
        v16bf bn = b1;
        if (j + 2 < 8) bn = load_bfrag(&lB[(j + 2) * 16 + br][0], kb2);
        acc[j] = __builtin_amdgcn_wmma_f32_16x16x32_bf16(
            false, a, false, b0, (short)0, acc[j], false, false);
        b0 = b1;
        b1 = bn;
    }
}

__global__ __launch_bounds__(256) void k_gemm_gates(
        const __bf16* __restrict__ xb, const __bf16* __restrict__ wt,
        const float* __restrict__ b_f, const float* __restrict__ b_i,
        const float* __restrict__ b_ig, const float* __restrict__ b_og,
        float* __restrict__ f_arr, float* __restrict__ z_arr,
        float* __restrict__ out) {
    __shared__ alignas(16) __bf16 ldsA0[128][40], ldsA1[128][40];
    __shared__ alignas(16) __bf16 ldsB0[128][40], ldsB1[128][40];

    const int tid  = threadIdx.x;
    const int lane = tid & 31;
    const int w    = tid >> 5;
    const int nBase = blockIdx.x * 32;
    const int mBase = blockIdx.y * 128;

    // per-thread staging geometry (2 x 16B chunks each for A and B)
    const int c0 = tid,        r0 = c0 >> 2, o0 = (c0 & 3) * 8;
    const int c1 = tid + 256,  r1 = c1 >> 2, o1 = (c1 & 3) * 8;
    const __bf16* pA0 = xb + (size_t)(mBase + r0) * DIM + o0;
    const __bf16* pA1 = xb + (size_t)(mBase + r1) * DIM + o1;
    const __bf16* pB0 = wt + (size_t)((r0 >> 5) * DIM + nBase + (r0 & 31)) * DIM + o0;
    const __bf16* pB1 = wt + (size_t)((r1 >> 5) * DIM + nBase + (r1 & 31)) * DIM + o1;

    const int kb  = (lane >> 4) << 3;   // unused: A and B frags both use kb2
    const int kb2 = (lane >> 4) << 4;   // B frag K-group base
    const int ar  = w * 16 + (lane & 15);
    const int br  = lane & 15;

    v8f acc[8];   // [g*2 + nt]
#pragma unroll
    for (int j = 0; j < 8; ++j) acc[j] = (v8f)(0.0f);

    v8bf sA0, sA1, sB0, sB1;
#define GLB_LOAD(k0)                                                     \
    do {                                                                 \
        sA0 = *(const v8bf*)(pA0 + (k0));                                \
        sA1 = *(const v8bf*)(pA1 + (k0));                                \
        sB0 = *(const v8bf*)(pB0 + (k0));                                \
        sB1 = *(const v8bf*)(pB1 + (k0));                                \
        SCHED_FENCE();  /* keep global loads issued before the wmmas */  \
    } while (0)
#define LDS_STORE(LA, LB)                                                \
    do {                                                                 \
        *(v8bf*)&LA[r0][o0] = sA0;                                       \
        *(v8bf*)&LA[r1][o1] = sA1;                                       \
        *(v8bf*)&LB[r0][o0] = sB0;                                       \
        *(v8bf*)&LB[r1][o1] = sB1;                                       \
    } while (0)

    // prologue: stage K-step 0 into buffer 0
    GLB_LOAD(0);
    LDS_STORE(ldsA0, ldsB0);

    // 15 double-steps cover it = 0..29 (even: buf0->buf1, odd: buf1->buf0)
#pragma unroll 1
    for (int it2 = 0; it2 < 15; ++it2) {
        const int it = it2 * 2;
        __syncthreads();
        GLB_LOAD((it + 1) * 32);
        compute_tile(ldsA0, ldsB0, ar, br, kb, kb2, acc);
        LDS_STORE(ldsA1, ldsB1);
        SCHED_TILE_FULL();

        __syncthreads();
        GLB_LOAD((it + 2) * 32);
        compute_tile(ldsA1, ldsB1, ar, br, kb, kb2, acc);
        LDS_STORE(ldsA0, ldsB0);
        SCHED_TILE_FULL();
    }
    // it = 30: compute buf0, stage final K-step into buf1
    __syncthreads();
    GLB_LOAD(31 * 32);
    compute_tile(ldsA0, ldsB0, ar, br, kb, kb2, acc);
    LDS_STORE(ldsA1, ldsB1);
    SCHED_TILE_FULL();
    // it = 31: final compute
    __syncthreads();
    compute_tile(ldsA1, ldsB1, ar, br, kb, kb2, acc);
    SCHED_TILE_COMPUTE();

#undef GLB_LOAD
#undef LDS_STORE

    // epilogue: this thread holds all 4 gates at identical (m, col) positions
    const int mrow0 = mBase + w * 16 + ((lane >> 4) << 3);
#pragma unroll
    for (int nt = 0; nt < 2; ++nt) {
        const int col = nBase + nt * 16 + (lane & 15);
        const float bfv = b_f[col], biv = b_i[col];
        const float bigv = b_ig[col], bogv = b_og[col];
#pragma unroll
        for (int r = 0; r < 8; ++r) {
            size_t idx = (size_t)(mrow0 + r) * DIM + col;
            float pf  = acc[0 * 2 + nt][r] + bfv;
            float pi  = acc[1 * 2 + nt][r] + biv;
            float pig = acc[2 * 2 + nt][r] + bigv;
            float pog = acc[3 * 2 + nt][r] + bogv;
            f_arr[idx] = sigmoidf_(pf);                    // forget gate
            z_arr[idx] = tanhf(pi) * sigmoidf_(pig);       // input term
            out[idx]   = sigmoidf_(pog);                   // output gate (temp in d_out)
        }
    }
}

// ---------------- chunked scan, pass 1: per-chunk (a = prod f, b = local end)
__global__ __launch_bounds__(256) void k_scan1(const float* __restrict__ f_arr,
                                               const float* __restrict__ z_arr,
                                               float* __restrict__ Ach,
                                               float* __restrict__ Bch) {
    const int bid  = blockIdx.x;          // ((b*NC + c)*4 + dblk)
    const int dblk = bid & 3;
    const int c    = (bid >> 2) & (NC - 1);
    const int b    = bid >> 9;
    const int d    = dblk * 256 + threadIdx.x;
    float a = 1.0f, h = 0.0f;
    const int base = (b * SEQ + c * CL) * DIM + d;
#pragma unroll 4
    for (int s = 0; s < CL; ++s) {
        float fv = f_arr[base + s * DIM];
        float zv = z_arr[base + s * DIM];
        a *= fv;
        h = fv * h + zv;
    }
    const int ci = (b * NC + c) * DIM + d;
    Ach[ci] = a;
    Bch[ci] = h;
}

// ---------------- pass 2: inter-chunk scan (128 steps, 4096 chains) ---------
__global__ __launch_bounds__(256) void k_scan2(const float* __restrict__ Ach,
                                               const float* __restrict__ Bch,
                                               const float* __restrict__ hinit,
                                               float* __restrict__ Carry) {
    const int t = blockIdx.x * 256 + threadIdx.x;   // 0..4095
    const int b = t >> 10, d = t & (DIM - 1);
    float carry = hinit[d];                          // h_{-1} seed
    for (int c = 0; c < NC; ++c) {
        const int ci = (b * NC + c) * DIM + d;
        Carry[ci] = carry;                           // carry entering chunk c
        carry = Ach[ci] * carry + Bch[ci];
    }
}

// ---------------- pass 3: apply carries, fuse output nonlinearity -----------
__global__ __launch_bounds__(256) void k_scan3(const float* __restrict__ f_arr,
                                               const float* __restrict__ z_arr,
                                               const float* __restrict__ Carry,
                                               float* __restrict__ out) {
    const int bid  = blockIdx.x;
    const int dblk = bid & 3;
    const int c    = (bid >> 2) & (NC - 1);
    const int b    = bid >> 9;
    const int d    = dblk * 256 + threadIdx.x;
    float h = Carry[(b * NC + c) * DIM + d];
    const int base = (b * SEQ + c * CL) * DIM + d;
#pragma unroll 4
    for (int s = 0; s < CL; ++s) {
        const int idx = base + s * DIM;
        float fv = f_arr[idx];
        float zv = z_arr[idx];
        h = fv * h + zv;
        float gv = out[idx];                 // sigmoid(og) staged by GEMM
        out[idx] = tanhf(h) * gv;            // final y
    }
}

extern "C" void kernel_launch(void* const* d_in, const int* in_sizes, int n_in,
                              void* d_out, int out_size, void* d_ws, size_t ws_size,
                              hipStream_t stream) {
    const float* x     = (const float*)d_in[0];
    const float* W_f   = (const float*)d_in[1];
    const float* b_f   = (const float*)d_in[2];
    const float* W_i   = (const float*)d_in[3];
    const float* b_i   = (const float*)d_in[4];
    const float* W_ig  = (const float*)d_in[5];
    const float* b_ig  = (const float*)d_in[6];
    const float* W_og  = (const float*)d_in[7];
    const float* b_og  = (const float*)d_in[8];
    const float* hinit = (const float*)d_in[9];
    float* out = (float*)d_out;

    char* p = (char*)d_ws;
    __bf16* xb = (__bf16*)p;  p += (size_t)BL * DIM * 2;            //  64 MB
    __bf16* wt = (__bf16*)p;  p += (size_t)NGATE * DIM * DIM * 2;   //   8 MB
    float* f_arr = (float*)p; p += (size_t)BL * DIM * 4;            // 128 MB
    float* z_arr = (float*)p; p += (size_t)BL * DIM * 4;            // 128 MB
    float* Ach   = (float*)p; p += (size_t)BATCH * NC * DIM * 4;    //   2 MB
    float* Bch   = (float*)p; p += (size_t)BATCH * NC * DIM * 4;    //   2 MB
    float* Carry = (float*)p;                                       //   2 MB

    k_cvt_x<<<(size_t)BL * DIM / 1024, 256, 0, stream>>>(x, xb);
    k_tr_w<<<dim3(32, 32, NGATE), 256, 0, stream>>>(W_f, W_i, W_ig, W_og, wt);
    k_gemm_gates<<<dim3(DIM / 32, BL / 128), 256, 0, stream>>>(
        xb, wt, b_f, b_i, b_ig, b_og, f_arr, z_arr, out);
    k_scan1<<<BATCH * NC * 4, 256, 0, stream>>>(f_arr, z_arr, Ach, Bch);
    k_scan2<<<16, 256, 0, stream>>>(Ach, Bch, hinit, Carry);
    k_scan3<<<BATCH * NC * 4, 256, 0, stream>>>(f_arr, z_arr, Carry, out);
}
